// Attention_36438502539877
// MI455X (gfx1250) — matrix operand order repaired and checked by reading, not compiled, not run on hardware
//
#include <hip/hip_runtime.h>

typedef _Float16 v16h __attribute__((ext_vector_type(16)));
typedef _Float16 v8h  __attribute__((ext_vector_type(8)));
typedef float    v8f  __attribute__((ext_vector_type(8)));

#define LOG2E 1.44269504088896340736f

#if __has_builtin(__builtin_amdgcn_exp2f)
#define EXP2F(x) __builtin_amdgcn_exp2f(x)
#else
#define EXP2F(x) exp2f(x)
#endif

#if __has_builtin(__builtin_amdgcn_rcpf)
#define RCPF(x) __builtin_amdgcn_rcpf(x)
#else
#define RCPF(x) (1.0f / (x))
#endif

#if __has_builtin(__builtin_amdgcn_s_wait_asynccnt)
#define WAIT_ASYNC(n) __builtin_amdgcn_s_wait_asynccnt(n)
#else
#define WAIT_ASYNC(n) asm volatile("s_wait_asynccnt %0" ::"i"(n) : "memory")
#endif

// Async DMA: copy 16 bytes/lane from global memory into LDS (ASYNCcnt).
__device__ __forceinline__ void async_load_b128(unsigned lds_off,
                                                const void* gptr) {
  asm volatile("global_load_async_to_lds_b128 %0, %1, off" ::"v"(lds_off),
               "v"(gptr)
               : "memory");
}

__device__ __forceinline__ v8f wmma16(v16h a, v16h b, v8f c) {
  // D = A(16x32 f16) x B(32x16 f16) + C(16x16 f32)
  return __builtin_amdgcn_wmma_f32_16x16x32_f16(false, a, false, b, (short)0, c,
                                                false, false);
}

// ds_swizzle bitmask mode: and_mask[4:0]=0x1f, or_mask[9:5]=0, xor_mask[14:10]=M
template <int M>
__device__ __forceinline__ float swz_xor(float x) {
  return __int_as_float(
      __builtin_amdgcn_ds_swizzle(__float_as_int(x), (M << 10) | 0x1f));
}
__device__ __forceinline__ float redmax16(float t) {
  t = fmaxf(t, swz_xor<1>(t));
  t = fmaxf(t, swz_xor<2>(t));
  t = fmaxf(t, swz_xor<4>(t));
  t = fmaxf(t, swz_xor<8>(t));
  return t;
}
__device__ __forceinline__ float redsum16(float t) {
  t += swz_xor<1>(t);
  t += swz_xor<2>(t);
  t += swz_xor<4>(t);
  t += swz_xor<8>(t);
  return t;
}

// B operand (32x16, K x N), f16: lane holds column N = lane&15; VGPRs hold
// 16 consecutive K halves; lanes 16-31 take K = 16..31. One contiguous 32B load.
__device__ __forceinline__ v16h load_b16g(const _Float16* __restrict__ base,
                                          int strideHalves, int lane) {
  return *(const v16h*)(base + (size_t)(lane & 15) * strideHalves +
                        (lane >> 4) * 16);
}

// A operand (16x32, M x K), f16, from a row-major [16][32] tile:
// lane L: M=L&15; v0-7 hold K = h*16..h*16+15
__device__ __forceinline__ v16h load_a16h(const _Float16* base, int lane) {
  const _Float16* p = base + (lane & 15) * 32 + (lane >> 4) * 16;
  v8h lo = *(const v8h*)(p);
  v8h hi = *(const v8h*)(p + 8);
  v16h a;
#pragma unroll
  for (int i = 0; i < 8; ++i) { a[i] = lo[i]; a[i + 8] = hi[i]; }
  return a;
}

// ---------------------------------------------------------------------------
// Kernel 1: convert/transpose weights to f16 for WMMA B-operand layouts.
__global__ __launch_bounds__(256) void prep_weights(
    const float* __restrict__ Wf, const float* __restrict__ Wg,
    const float* __restrict__ Wh, const float* __restrict__ Wv,
    _Float16* __restrict__ Wft, _Float16* __restrict__ Wgt,
    _Float16* __restrict__ Wht, _Float16* __restrict__ Wvt) {
  int i = blockIdx.x * 256 + threadIdx.x;  // 0..8191
  int k = i >> 5, n = i & 31;              // Wf[k][n]
  Wft[n * 256 + k] = (_Float16)Wf[i];
  Wgt[n * 256 + k] = (_Float16)Wg[i];
  Wht[n * 256 + k] = (_Float16)Wh[i];
  int k2 = i >> 8, n2 = i & 255;           // Wv[k2][n2]
  Wvt[n2 * 32 + k2] = (_Float16)Wv[i];
}

// ---------------------------------------------------------------------------
// Kernel 2: fused f/g/h projection. x[16384][256] f32 -> f16 f,g (row-major
// [16384][32]) and h transposed per batch ([4][32][4096]).
__global__ __launch_bounds__(128) void proj_kernel(
    const float* __restrict__ x, const _Float16* __restrict__ Wft,
    const _Float16* __restrict__ Wgt, const _Float16* __restrict__ Wht,
    const float* __restrict__ bf, const float* __restrict__ bg,
    const float* __restrict__ bh, _Float16* __restrict__ fbuf,
    _Float16* __restrict__ gbuf, _Float16* __restrict__ htb) {
  int lane = threadIdx.x & 31;
  int w = threadIdx.x >> 5;
  int R = (blockIdx.x * 4 + w) * 16;
  int c0 = lane & 15, hh = lane >> 4;

  v8f aF0 = {}, aF1 = {}, aG0 = {}, aG1 = {}, aH0 = {}, aH1 = {};
  const float* xrow = x + (size_t)(R + c0) * 256 + hh * 16;

#pragma unroll
  for (int ks = 0; ks < 8; ++ks) {
    int k0 = ks * 32;
    const float* p = xrow + k0;
    float4 q0 = *(const float4*)(p);
    float4 q1 = *(const float4*)(p + 4);
    float4 q2 = *(const float4*)(p + 8);
    float4 q3 = *(const float4*)(p + 12);
    v16h a;
    a[0]  = (_Float16)q0.x; a[1]  = (_Float16)q0.y;
    a[2]  = (_Float16)q0.z; a[3]  = (_Float16)q0.w;
    a[4]  = (_Float16)q1.x; a[5]  = (_Float16)q1.y;
    a[6]  = (_Float16)q1.z; a[7]  = (_Float16)q1.w;
    a[8]  = (_Float16)q2.x; a[9]  = (_Float16)q2.y;
    a[10] = (_Float16)q2.z; a[11] = (_Float16)q2.w;
    a[12] = (_Float16)q3.x; a[13] = (_Float16)q3.y;
    a[14] = (_Float16)q3.z; a[15] = (_Float16)q3.w;

    v16h bF0 = load_b16g(Wft + k0, 256, lane);
    v16h bF1 = load_b16g(Wft + 16 * 256 + k0, 256, lane);
    v16h bG0 = load_b16g(Wgt + k0, 256, lane);
    v16h bG1 = load_b16g(Wgt + 16 * 256 + k0, 256, lane);
    v16h bH0 = load_b16g(Wht + k0, 256, lane);
    v16h bH1 = load_b16g(Wht + 16 * 256 + k0, 256, lane);
    aF0 = wmma16(a, bF0, aF0);
    aF1 = wmma16(a, bF1, aF1);
    aG0 = wmma16(a, bG0, aG0);
    aG1 = wmma16(a, bG1, aG1);
    aH0 = wmma16(a, bH0, aH0);
    aH1 = wmma16(a, bH1, aH1);
  }

  float bfv0 = bf[c0], bfv1 = bf[c0 + 16];
  float bgv0 = bg[c0], bgv1 = bg[c0 + 16];
  float bhv0 = bh[c0], bhv1 = bh[c0 + 16];
#pragma unroll
  for (int r = 0; r < 8; ++r) {
    int row = R + r + hh * 8;  // C-layout: lanes 16-31 hold rows M+8
    fbuf[row * 32 + c0]      = (_Float16)(aF0[r] + bfv0);
    fbuf[row * 32 + 16 + c0] = (_Float16)(aF1[r] + bfv1);
    gbuf[row * 32 + c0]      = (_Float16)(aG0[r] + bgv0);
    gbuf[row * 32 + 16 + c0] = (_Float16)(aG1[r] + bgv1);
    int b = row >> 12, rl = row & 4095;
    htb[((b * 32) + c0) * 4096 + rl]      = (_Float16)(aH0[r] + bhv0);
    htb[((b * 32) + 16 + c0) * 4096 + rl] = (_Float16)(aH1[r] + bhv1);
  }
}

// ---------------------------------------------------------------------------
// Kernel 3: flash attention + output projection + residual, fully fused.
// One wave = 16 query rows; 4 waves per block; grid (64, B).
// f/h key blocks are staged into LDS once per block via async global->LDS DMA
// (triple-buffered, one barrier per iteration) and shared by all 4 waves.
__global__ __launch_bounds__(128) void attn_kernel(
    const float* __restrict__ x, const _Float16* __restrict__ fbuf,
    const _Float16* __restrict__ gbuf, const _Float16* __restrict__ htb,
    const _Float16* __restrict__ Wvt, const float* __restrict__ bv,
    const float* __restrict__ gammap, float* __restrict__ out) {
  __shared__ _Float16 pl[4][16 * 32];   // per-wave p-tile transpose buffer
  __shared__ _Float16 sf[3][32 * 32];   // staged f: [32 keys][32 ch]
  __shared__ _Float16 sh[3][32 * 32];   // staged h: [32 cr][32 keys]

  int lane = threadIdx.x & 31, w = threadIdx.x >> 5;
  int batch = blockIdx.y;
  int qbase = (blockIdx.x * 4 + w) * 16;
  int c0 = lane & 15, hh = lane >> 4;

  const _Float16* gb = gbuf + (size_t)(batch * 4096 + qbase) * 32;
  const char* fbB = (const char*)(fbuf + (size_t)batch * 4096 * 32);
  const char* hbB = (const char*)(htb + (size_t)batch * 32 * 4096);

  // LDS byte offsets (low 32 bits of the flat LDS address = LDS offset)
  unsigned sf_base = (unsigned)(uintptr_t)(&sf[0][0]);
  unsigned sh_base = (unsigned)(uintptr_t)(&sh[0][0]);

  v16h ga = load_a16h(gb, lane);  // resident query A-tile
  v8f acc0 = {}, acc1 = {};
  v8f zero = {};
  float m[8], l[8];
#pragma unroll
  for (int r = 0; r < 8; ++r) { m[r] = -3.0e38f; l[r] = 0.f; }

  // wave 0 stages one 32-key block (8 x async b128 = 4KB) into LDS slot.
  auto stage = [&](int slot, int kb) {
    if (w == 0) {
      unsigned fl = sf_base + (unsigned)slot * 2048u + lane * 16u;
      // f rows kb..kb+31 are 2KB contiguous
#pragma unroll
      for (int t = 0; t < 4; ++t)
        async_load_b128(fl + t * 512u, fbB + (size_t)kb * 64 + t * 512 + lane * 16);
      // h: 32 rows (cr), 64B each, global row stride 8192B
      int row = lane >> 2, seg = lane & 3;
#pragma unroll
      for (int t = 0; t < 4; ++t)
        async_load_b128(sh_base + (unsigned)slot * 2048u +
                            (unsigned)(t * 8 + row) * 64u + seg * 16u,
                        hbB + (size_t)(t * 8 + row) * 8192 + (size_t)kb * 2 +
                            seg * 16);
    }
  };

  stage(0, 0);
  stage(1, 32);

  for (int it = 0; it < 128; ++it) {
    int kb = it * 32;
    if (w == 0) WAIT_ASYNC(8);  // stage `it` landed (async loads retire in order)
    __syncthreads();            // publish slot it%3; slot (it+2)%3 now idle
    if (kb + 64 < 4096) stage((it + 2) % 3, kb + 64);

    int slot = it % 3;
    const _Float16* sfp = &sf[slot][0];
    const _Float16* shp = &sh[slot][0];

    v16h fb0 = *(const v16h*)(sfp + c0 * 32 + hh * 16);          // keys +0..15
    v16h fb1 = *(const v16h*)(sfp + (16 + c0) * 32 + hh * 16);   // keys +16..31
    v8f s0 = wmma16(ga, fb0, zero);
    v8f s1 = wmma16(ga, fb1, zero);

    float alpha[8];
#pragma unroll
    for (int r = 0; r < 8; ++r) {
      float t = redmax16(fmaxf(s0[r], s1[r]));
      float mn = fmaxf(m[r], t);
      alpha[r] = EXP2F((m[r] - mn) * LOG2E);
      m[r] = mn;
    }
#pragma unroll
    for (int r = 0; r < 8; ++r) {
      float p0 = EXP2F((s0[r] - m[r]) * LOG2E);
      float p1 = EXP2F((s1[r] - m[r]) * LOG2E);
      l[r] = l[r] * alpha[r] + redsum16(p0 + p1);
      acc0[r] *= alpha[r];
      acc1[r] *= alpha[r];
      int row = r + hh * 8;
      pl[w][row * 32 + c0]      = (_Float16)p0;
      pl[w][row * 32 + 16 + c0] = (_Float16)p1;
    }
    __builtin_amdgcn_wave_barrier();  // DS in-order per wave; fence compiler
    v16h pa = load_a16h(&pl[w][0], lane);
    __builtin_amdgcn_wave_barrier();

    v16h hb0 = *(const v16h*)(shp + c0 * 32 + hh * 16);          // cr 0..15
    v16h hb1 = *(const v16h*)(shp + (16 + c0) * 32 + hh * 16);   // cr 16..31
    acc0 = wmma16(pa, hb0, acc0);
    acc1 = wmma16(pa, hb1, acc1);
  }

  // v = acc / l  -> f16 A-tile via LDS
#pragma unroll
  for (int r = 0; r < 8; ++r) {
    float rl = RCPF(l[r]);
    int row = r + hh * 8;
    pl[w][row * 32 + c0]      = (_Float16)(acc0[r] * rl);
    pl[w][row * 32 + 16 + c0] = (_Float16)(acc1[r] * rl);
  }
  __builtin_amdgcn_wave_barrier();
  v16h va = load_a16h(&pl[w][0], lane);
  __builtin_amdgcn_wave_barrier();

  float gamma = gammap[0];
#pragma unroll 4
  for (int nt = 0; nt < 16; ++nt) {  // 256 output channels / 16
    v16h wb = load_b16g(Wvt + nt * 16 * 32, 32, lane);
    v8f o = wmma16(va, wb, zero);
    int col = nt * 16 + c0;
    float bias = bv[col];
#pragma unroll
    for (int r = 0; r < 8; ++r) {
      size_t idx = ((size_t)(batch * 4096 + qbase + r + hh * 8)) * 256 + col;
      out[idx] = gamma * (o[r] + bias) + x[idx];
    }
  }
}

// ---------------------------------------------------------------------------
extern "C" void kernel_launch(void* const* d_in, const int* in_sizes, int n_in,
                              void* d_out, int out_size, void* d_ws,
                              size_t ws_size, hipStream_t stream) {
  const float* x  = (const float*)d_in[0];
  const float* Wf = (const float*)d_in[1];
  const float* bf = (const float*)d_in[2];
  const float* Wg = (const float*)d_in[3];
  const float* bg = (const float*)d_in[4];
  const float* Wh = (const float*)d_in[5];
  const float* bh = (const float*)d_in[6];
  const float* Wv = (const float*)d_in[7];
  const float* bv = (const float*)d_in[8];
  const float* gm = (const float*)d_in[9];
  float* out = (float*)d_out;

  _Float16* fbuf = (_Float16*)d_ws;
  _Float16* gbuf = fbuf + 16384 * 32;
  _Float16* htb  = gbuf + 16384 * 32;
  _Float16* Wft  = htb + 4 * 32 * 4096;
  _Float16* Wgt  = Wft + 32 * 256;
  _Float16* Wht  = Wgt + 32 * 256;
  _Float16* Wvt  = Wht + 32 * 256;

  prep_weights<<<dim3(32), dim3(256), 0, stream>>>(Wf, Wg, Wh, Wv, Wft, Wgt,
                                                   Wht, Wvt);
  proj_kernel<<<dim3(256), dim3(128), 0, stream>>>(x, Wft, Wgt, Wht, bf, bg,
                                                   bh, fbuf, gbuf, htb);
  attn_kernel<<<dim3(64, 4), dim3(128), 0, stream>>>(x, fbuf, gbuf, htb, Wvt,
                                                     bv, gm, out);
}
